// GPT_3719441678497
// MI455X (gfx1250) — matrix operand order repaired and checked by reading, not compiled, not run on hardware
//
#include <hip/hip_runtime.h>
#include <hip/hip_bf16.h>
#include <math.h>

// Problem constants (match reference)
#define BB   2
#define TT   1024
#define DD   1024
#define HH   16
#define HDIM 64
#define LL   4
#define VV   32000
#define NBK  16
#define DFF  4096
#define EPSF 1.1920929e-07f
#define THRF 0.05f

typedef __attribute__((ext_vector_type(16))) __bf16 v16bf;
typedef __attribute__((ext_vector_type(8)))  float  v8f;
typedef __attribute__((ext_vector_type(4)))  int    v4i;
typedef __attribute__((ext_vector_type(8)))  int    v8i;

union FragBF {
  v16bf v;
  uint4 u[2];
};

__device__ __forceinline__ v8f wmma_bf16(const FragBF& a, const FragBF& b, v8f c) {
  return __builtin_amdgcn_wmma_f32_16x16x32_bf16(false, a.v, false, b.v, (short)0, c,
                                                 false, false);
}

// A-fragment (16x32, 16-bit): lane = M row (lane&15); per-lane 16 contiguous K at col+half*16.
__device__ __forceinline__ void loadA(FragBF& f, const __hip_bfloat16* rowPtr, int col, int half) {
  f.u[0] = *reinterpret_cast<const uint4*>(rowPtr + col + half * 16);
  f.u[1] = *reinterpret_cast<const uint4*>(rowPtr + col + half * 16 + 8);
}
// B-fragment (32x16, 16-bit): lane = N col (lane&15); per-lane 16 contiguous K at col+half*16.
__device__ __forceinline__ void loadB(FragBF& f, const __hip_bfloat16* rowPtr, int col, int half) {
  f.u[0] = *reinterpret_cast<const uint4*>(rowPtr + col + half * 16);
  f.u[1] = *reinterpret_cast<const uint4*>(rowPtr + col + half * 16 + 8);
}

// async global->LDS copy of one 16-byte chunk per lane (GV mode, ASYNCcnt)
__device__ __forceinline__ void asyncCopy16(unsigned ldsByteOff, const void* gsrc) {
  asm volatile("global_load_async_to_lds_b128 %0, %1, off"
               :: "v"(ldsByteOff), "v"(gsrc)
               : "memory");
}
__device__ __forceinline__ void waitAsync0() {
  asm volatile("s_wait_asynccnt 0x0" ::: "memory");
}
__device__ __forceinline__ void waitAsync8() {
  asm volatile("s_wait_asynccnt 0x8" ::: "memory");
}

// ---------------------------------------------------------------------------
// Generic bf16 WMMA GEMM: C[M,N] = epi(A[M,K] @ Wt[N,K]^T + bias) (+ residual)
// Tile 128x128 per block (256 threads = 8 waves, 2(M) x 4(N) wave grid).
// The 128x32 B (weight) tile is staged to LDS by the Tensor Data Mover:
// wave 0 issues one tensor_load_to_lds per k-step (2D descriptor in SGPRs),
// waits TENSORcnt, and a workgroup barrier publishes the tile to all 8 waves.
// A fragments use the direct pipelined global->VGPR path.
// M,N multiples of 128; K multiple of 32.
// ---------------------------------------------------------------------------
__global__ __launch_bounds__(256) void gemm_bf16_wmma(
    const __hip_bfloat16* __restrict__ A,
    const __hip_bfloat16* __restrict__ Wt,
    const float* __restrict__ bias,
    const float* __restrict__ residual,
    float* __restrict__ outF,
    __hip_bfloat16* __restrict__ outBF,
    int M, int N, int K, int epi,
    long long strideA, long long strideW, long long strideC)
{
  // double-buffered B tile: 2 x (128 rows x 32 cols) bf16 = 2 x 8KB
  __shared__ __align__(16) __hip_bfloat16 btile[2 * 128 * 32];

  const int tid  = threadIdx.x;
  const int lane = tid & 31;
  const int wave = tid >> 5;
  const int half = lane >> 4;
  const int lrow = lane & 15;
  const int wm = wave & 1, wn = wave >> 1;
  const int nblk = blockIdx.x * 128;
  const int m0 = blockIdx.y * 128 + wm * 64;
  const int n0 = nblk + wn * 32;
  const long long bofA = (long long)blockIdx.z * strideA;
  const long long bofW = (long long)blockIdx.z * strideW;
  const long long bofC = (long long)blockIdx.z * strideC;

  const unsigned ldsBase = (unsigned)(size_t)&btile[0];
  const __hip_bfloat16* WtB = Wt + bofW;

  // TDM descriptor group 1 (k-invariant): data_size=4B, tile 16dw x 128 rows,
  // tensor row length/stride = K/2 dwords.
  const int td0 = K >> 1;
  v8i g1;
  g1[0] = 2 << 16;                                   // data_size=2 (4 bytes)
  g1[1] = (td0 & 0xFFFF) << 16;                      // tensor_dim0 lo16 -> [31:16]
  g1[2] = ((td0 >> 16) & 0xFFFF) | (128 << 16);      // tensor_dim0 hi16 | tensor_dim1 lo16
  g1[3] = 16 << 16;                                  // tensor_dim1 hi16=0 | tile_dim0=16 dw
  g1[4] = 128;                                       // tile_dim1=128, tile_dim2=0
  g1[5] = td0;                                       // tensor_dim0_stride lo32 (dwords)
  g1[6] = 0;                                         // stride0 hi16 | stride1 lo16
  g1[7] = 0;                                         // stride1 hi32

  auto stageTDM = [&](int buf, int k) {
    if (wave == 0) {
      const unsigned long long ga =
          (unsigned long long)(size_t)(WtB + (long long)nblk * K + k);
      v4i g0;
      g0[0] = 1;                                         // count=1, user descriptor
      g0[1] = (int)(ldsBase + (unsigned)(buf * 8192));   // lds_addr (bytes)
      g0[2] = (int)(unsigned)(ga & 0xFFFFFFFFull);       // global_addr lo32
      g0[3] = (int)((unsigned)((ga >> 32) & 0x1FFFFFFu)  // global_addr [56:32]
                    | 0x80000000u);                      // type=2 ("image")
      asm volatile("tensor_load_to_lds %0, %1" :: "s"(g0), "s"(g1) : "memory");
    }
  };

  const __hip_bfloat16* aRow[4];
#pragma unroll
  for (int i = 0; i < 4; ++i)
    aRow[i] = A + bofA + (long long)(m0 + i * 16 + lrow) * K;

  // prologue: stage first tile
  stageTDM(0, 0);
  if (wave == 0) __builtin_amdgcn_s_wait_tensorcnt(0);
  __syncthreads();

  v8f acc[4][2] = {};
  int buf = 0;
  for (int k = 0; k < K; k += 32) {
    const bool more = (k + 32) < K;
    if (more) stageTDM(buf ^ 1, k + 32);   // TDM overlaps the WMMAs below

    FragBF a[4], b[2];
#pragma unroll
    for (int i = 0; i < 4; ++i) loadA(a[i], aRow[i], k, half);
#pragma unroll
    for (int j = 0; j < 2; ++j) {
      const __hip_bfloat16* bsrc =
          &btile[buf * 4096 + (wn * 32 + j * 16 + lrow) * 32];
      b[j].u[0] = *reinterpret_cast<const uint4*>(bsrc + half * 16);
      b[j].u[1] = *reinterpret_cast<const uint4*>(bsrc + half * 16 + 8);
    }
    __builtin_prefetch(aRow[0] + k + 256, 0, 1);
    __builtin_prefetch(aRow[2] + k + 256, 0, 1);
#pragma unroll
    for (int i = 0; i < 4; ++i)
#pragma unroll
      for (int j = 0; j < 2; ++j)
        acc[i][j] = wmma_bf16(a[i], b[j], acc[i][j]);

    if (more) {
      if (wave == 0) __builtin_amdgcn_s_wait_tensorcnt(0);
      __syncthreads();
      buf ^= 1;
    }
  }

  // Epilogue. C layout: vgpr r -> row m0 + i*16 + r + 8*half ; col = n0 + j*16 + (lane&15)
#pragma unroll
  for (int i = 0; i < 4; ++i) {
#pragma unroll
    for (int j = 0; j < 2; ++j) {
      const int nn = n0 + j * 16 + lrow;
      const float bv = bias ? bias[nn] : 0.0f;
#pragma unroll
      for (int r = 0; r < 8; ++r) {
        const int mm = m0 + i * 16 + r + 8 * half;
        float vv = acc[i][j][r] + bv;
        if (epi == 1)      vv = vv / (1.0f + __expf(-vv)); // SiLU
        else if (epi == 2) vv = tanhf(vv);
        const long long idx = bofC + (long long)mm * N + nn;
        if (residual) vv += residual[idx];
        if (outF)  outF[idx]  = vv;
        if (outBF) outBF[idx] = __float2bfloat16(vv);
      }
    }
  }
}

// ---------------------------------------------------------------------------
// Flash-style attention with WMMA. grid (T/64, B*H), block 128 (4 waves),
// each wave owns a 16-row q tile; s processed in chunks of 32.
// K chunks are double-buffered into wave-private LDS via async global->LDS
// copies; ASYNCcnt is per-wave so no barriers are needed — partial waits
// (s_wait_asynccnt 0x8) keep the next chunk's copies in flight.
// ---------------------------------------------------------------------------
__global__ __launch_bounds__(128) void attn_wmma(
    const __hip_bfloat16* __restrict__ q,   // [B*T, D]
    const __hip_bfloat16* __restrict__ k,   // [B*T, D]
    const __hip_bfloat16* __restrict__ vT,  // [B*H*HD, T]
    const float* __restrict__ sim,          // [B, T, T]
    __hip_bfloat16* __restrict__ o)         // [B*T, D]
{
  const int lane = threadIdx.x & 31;
  const int wave = threadIdx.x >> 5;
  const int half = lane >> 4;
  const int lrow = lane & 15;
  const int bh = blockIdx.y;
  const int b = bh / HH, h = bh % HH;
  const int qt = blockIdx.x * 64 + wave * 16;

  __shared__ __align__(16) __hip_bfloat16 pbuf[4][16 * 32];
  // per-wave double-buffered K tile: 32 rows x 64 bf16 = 4KB per buffer
  __shared__ __align__(16) __hip_bfloat16 kbuf[4][2][32 * 64];

  const unsigned kbufBase = (unsigned)(size_t)&kbuf[wave][0][0];

  auto stageK = [&](int kb, int sb) {
    // 4KB tile: 32 rows x 128B; each lane async-copies 8 x 16B chunks
#pragma unroll
    for (int c = 0; c < 8; ++c) {
      const int byteIdx = (lane + c * 32) * 16;  // 0..4095
      const int row  = byteIdx >> 7;             // 128B per row
      const int colE = (byteIdx & 127) >> 1;     // bf16 element within row
      const __hip_bfloat16* src =
          k + (long long)(b * TT + sb + row) * DD + h * HDIM + colE;
      asyncCopy16(kbufBase + (unsigned)(kb * 4096 + byteIdx), (const void*)src);
    }
  };

  const __hip_bfloat16* qRow = q + (long long)(b * TT + qt + lrow) * DD + h * HDIM;
  FragBF qf[2];
  loadA(qf[0], qRow, 0, half);
  loadA(qf[1], qRow, 32, half);

  v8f oacc[4] = {};
  float rm[8], rl[8];
#pragma unroll
  for (int r = 0; r < 8; ++r) { rm[r] = -INFINITY; rl[r] = 0.0f; }

  const float scale = 0.125f; // HD^-0.5 = 1/8
  const float* simRowBase = sim + (long long)b * TT * TT;

  stageK(0, 0);  // prologue: 8 async copies in flight
  int kbi = 0;
  for (int sb = 0; sb < qt + 16; sb += 32) {
    const bool more = (sb + 32) < (qt + 16);
    if (more) stageK(kbi ^ 1, sb + 32);  // issue next chunk (8 more in flight)
    if (more) waitAsync8();              // current chunk resident, next in flight
    else      waitAsync0();

    // S = Q K^T from LDS-resident K tile
    v8f s[2] = {};
#pragma unroll
    for (int j = 0; j < 2; ++j) {
      const __hip_bfloat16* kRow = &kbuf[wave][kbi][(j * 16 + lrow) * 64];
      FragBF kb0, kb1;
      loadB(kb0, kRow, 0, half);
      loadB(kb1, kRow, 32, half);
      s[j] = wmma_bf16(qf[0], kb0, s[j]);
      s[j] = wmma_bf16(qf[1], kb1, s[j]);
    }
    // mask: causal && sim > THR
#pragma unroll
    for (int j = 0; j < 2; ++j) {
      const int nn = sb + j * 16 + lrow;
#pragma unroll
      for (int r = 0; r < 8; ++r) {
        const int mm = qt + r + 8 * half;
        const float sv = simRowBase[(long long)mm * TT + nn];
        const bool ok = (nn <= mm) && (sv > THRF);
        s[j][r] = ok ? s[j][r] * scale : -INFINITY;
      }
    }
    // online softmax: rows live across 16-lane half-groups
    float corr[8];
#pragma unroll
    for (int r = 0; r < 8; ++r) {
      float mx = fmaxf(s[0][r], s[1][r]);
      mx = fmaxf(mx, __shfl_xor(mx, 1, 16));
      mx = fmaxf(mx, __shfl_xor(mx, 2, 16));
      mx = fmaxf(mx, __shfl_xor(mx, 4, 16));
      mx = fmaxf(mx, __shfl_xor(mx, 8, 16));
      const float nm = fmaxf(rm[r], mx);
      const float p0 = (s[0][r] == -INFINITY) ? 0.0f : __expf(s[0][r] - nm);
      const float p1 = (s[1][r] == -INFINITY) ? 0.0f : __expf(s[1][r] - nm);
      float sum = p0 + p1;
      sum += __shfl_xor(sum, 1, 16);
      sum += __shfl_xor(sum, 2, 16);
      sum += __shfl_xor(sum, 4, 16);
      sum += __shfl_xor(sum, 8, 16);
      const float c = (nm == -INFINITY) ? 1.0f : __expf(rm[r] - nm);
      rl[r] = rl[r] * c + sum;
      rm[r] = nm;
      corr[r] = c;
      s[0][r] = p0;
      s[1][r] = p1;
    }
#pragma unroll
    for (int t = 0; t < 4; ++t)
#pragma unroll
      for (int r = 0; r < 8; ++r) oacc[t][r] *= corr[r];

    // stage P (C layout) through LDS, reload in A-fragment layout
#pragma unroll
    for (int j = 0; j < 2; ++j)
#pragma unroll
      for (int r = 0; r < 8; ++r)
        pbuf[wave][(r + 8 * half) * 32 + j * 16 + lrow] = __float2bfloat16(s[j][r]);
    FragBF pf;
    pf.u[0] = *reinterpret_cast<const uint4*>(&pbuf[wave][lrow * 32 + half * 16]);
    pf.u[1] = *reinterpret_cast<const uint4*>(&pbuf[wave][lrow * 32 + half * 16 + 8]);

    // O += P @ V  (4 hd n-tiles, K = 32 s-chunk)
#pragma unroll
    for (int t = 0; t < 4; ++t) {
      const __hip_bfloat16* vRow =
          vT + (long long)(bh * HDIM + t * 16 + lrow) * TT;
      FragBF vb;
      loadB(vb, vRow, sb, half);
      oacc[t] = wmma_bf16(pf, vb, oacc[t]);
    }
    kbi ^= 1;
  }

  // normalize and store O (bf16) back into [B*T, D]
#pragma unroll
  for (int r = 0; r < 8; ++r) {
    const float inv = rl[r] > 0.0f ? 1.0f / rl[r] : 0.0f;
    const int mm = qt + r + 8 * half;
#pragma unroll
    for (int t = 0; t < 4; ++t) {
      const int col = h * HDIM + t * 16 + lrow;
      o[(long long)(b * TT + mm) * DD + col] = __float2bfloat16(oacc[t][r] * inv);
    }
  }
}

// ---------------------------------------------------------------------------
// RMSNorm (fp32 in, bf16*w out). One 256-thread block per row.
// ---------------------------------------------------------------------------
__global__ __launch_bounds__(256) void rmsnorm_bf16(
    const float* __restrict__ x, const float* __restrict__ w,
    __hip_bfloat16* __restrict__ out)
{
  const int row = blockIdx.x;
  const int tid = threadIdx.x;
  const float* xr = x + (long long)row * DD;
  float ss = 0.0f;
  for (int i = tid; i < DD; i += 256) { const float v = xr[i]; ss += v * v; }
#pragma unroll
  for (int off = 1; off < 32; off <<= 1) ss += __shfl_xor(ss, off, 32);
  __shared__ float red[8];
  if ((tid & 31) == 0) red[tid >> 5] = ss;
  __syncthreads();
  float tot = 0.0f;
#pragma unroll
  for (int i = 0; i < 8; ++i) tot += red[i];
  const float sc = rsqrtf(tot / (float)DD + EPSF);
  __hip_bfloat16* orow = out + (long long)row * DD;
  for (int i = tid; i < DD; i += 256) orow[i] = __float2bfloat16(xr[i] * sc * w[i]);
}

// ---------------------------------------------------------------------------
// Embedding: x = token_emb[tok] + pos_emb
// ---------------------------------------------------------------------------
__global__ void embed_k(const int* __restrict__ tok, const float* __restrict__ emb,
                        const float* __restrict__ pos, float* __restrict__ x)
{
  const long long i = (long long)blockIdx.x * blockDim.x + threadIdx.x;
  if (i >= (long long)BB * TT * DD) return;
  const int d = (int)(i % DD);
  const long long bt = i / DD;
  const int t = (int)(bt % TT);
  x[i] = emb[(long long)tok[bt] * DD + d] + pos[(long long)t * DD + d];
}

// fp32 -> bf16 elementwise
__global__ void cvt_bf16_k(const float* __restrict__ in,
                           __hip_bfloat16* __restrict__ out, long long n)
{
  const long long i = (long long)blockIdx.x * blockDim.x + threadIdx.x;
  if (i < n) out[i] = __float2bfloat16(in[i]);
}

// fp32 [R,C] -> bf16 [C,R]  (R,C multiples of 32)
__global__ __launch_bounds__(256) void tcvt_k(const float* __restrict__ in,
                                              __hip_bfloat16* __restrict__ out,
                                              int R, int C)
{
  __shared__ float tile[32][33];
  const int c0 = blockIdx.x * 32, r0 = blockIdx.y * 32;
  const int tx = threadIdx.x, ty = threadIdx.y; // block (32,8)
#pragma unroll
  for (int i = 0; i < 4; ++i)
    tile[ty + i * 8][tx] = in[(long long)(r0 + ty + i * 8) * C + c0 + tx];
  __syncthreads();
#pragma unroll
  for (int i = 0; i < 4; ++i)
    out[(long long)(c0 + ty + i * 8) * R + r0 + tx] =
        __float2bfloat16(tile[tx][ty + i * 8]);
}

// v [B*T, D] (head-major cols) -> vT [B*H*HD, T]
__global__ void vtrans_k(const __hip_bfloat16* __restrict__ v,
                         __hip_bfloat16* __restrict__ vT)
{
  const long long i = (long long)blockIdx.x * blockDim.x + threadIdx.x;
  if (i >= (long long)BB * HH * HDIM * TT) return;
  const int t = (int)(i % TT);
  const long long r = i / TT;
  const int hd = (int)(r % HDIM);
  const long long bhh = r / HDIM;
  const int h = (int)(bhh % HH);
  const int b = (int)(bhh / HH);
  vT[i] = v[(long long)(b * TT + t) * DD + h * HDIM + hd];
}

// routing tail: logits = r1 @ rw2 + rb2 ; softmax over NB=16 -> probs bf16 [M,32] (padded)
__global__ __launch_bounds__(256) void routing_k(
    const __hip_bfloat16* __restrict__ r1, // [M,128], already tanh'd
    const float* __restrict__ rw2,         // [128,16]
    const float* __restrict__ rb2,         // [16]
    __hip_bfloat16* __restrict__ probs)    // [M,32] (cols 16..31 zero)
{
  const int tid = threadIdx.x;
  const int localTok = tid >> 4, bucket = tid & 15;
  const int tok = blockIdx.x * 16 + localTok;
  float acc = rb2[bucket];
  const __hip_bfloat16* rr = r1 + (long long)tok * 128;
  for (int kk = 0; kk < 128; ++kk)
    acc += __bfloat162float(rr[kk]) * rw2[kk * NBK + bucket];
  __shared__ float lg[16][16];
  __shared__ float ex[16][16];
  lg[localTok][bucket] = acc;
  __syncthreads();
  float mx = -INFINITY;
#pragma unroll
  for (int i = 0; i < 16; ++i) mx = fmaxf(mx, lg[localTok][i]);
  const float e = __expf(acc - mx);
  ex[localTok][bucket] = e;
  __syncthreads();
  float ssum = 0.0f;
#pragma unroll
  for (int i = 0; i < 16; ++i) ssum += ex[localTok][i];
  probs[(long long)tok * 32 + bucket] = __float2bfloat16(e / ssum);
  probs[(long long)tok * 32 + 16 + bucket] = __float2bfloat16(0.0f);
}

// ---------------------------------------------------------------------------
extern "C" void kernel_launch(void* const* d_in, const int* in_sizes, int n_in,
                              void* d_out, int out_size, void* d_ws, size_t ws_size,
                              hipStream_t stream)
{
  (void)in_sizes; (void)n_in; (void)out_size; (void)ws_size;

  const int*   tokens    = (const int*)  d_in[0];
  const float* token_emb = (const float*)d_in[1];
  const float* pos_emb   = (const float*)d_in[2];
  const float* Wq        = (const float*)d_in[3];
  const float* Wk        = (const float*)d_in[4];
  const float* Wv        = (const float*)d_in[5];
  const float* Wo        = (const float*)d_in[6];
  const float* rw1       = (const float*)d_in[7];
  const float* rb1       = (const float*)d_in[8];
  const float* rw2       = (const float*)d_in[9];
  const float* rb2       = (const float*)d_in[10];
  const float* n1w       = (const float*)d_in[11];
  const float* n2w       = (const float*)d_in[12];
  const float* mw1       = (const float*)d_in[13];
  const float* mb1       = (const float*)d_in[14];
  const float* mw2       = (const float*)d_in[15];
  const float* mb2       = (const float*)d_in[16];
  float* out = (float*)d_out;

  const long long MTOK = (long long)BB * TT; // 2048

  char* wp = (char*)d_ws;
  auto carve = [&](size_t bytes) -> void* {
    void* p = (void*)wp;
    wp += (bytes + 255) & ~(size_t)255;
    return p;
  };
  float*          xF     = (float*)carve(MTOK * DD * 4);
  __hip_bfloat16* hBF    = (__hip_bfloat16*)carve(MTOK * DD * 2);
  __hip_bfloat16* qBF    = (__hip_bfloat16*)carve(MTOK * DD * 2);
  __hip_bfloat16* kBF    = (__hip_bfloat16*)carve(MTOK * DD * 2);
  __hip_bfloat16* vBF    = (__hip_bfloat16*)carve(MTOK * DD * 2);
  __hip_bfloat16* vTbuf  = (__hip_bfloat16*)carve(MTOK * DD * 2);
  __hip_bfloat16* oBF    = (__hip_bfloat16*)carve(MTOK * DD * 2);
  __hip_bfloat16* midBF  = (__hip_bfloat16*)carve(MTOK * DFF * 2);
  __hip_bfloat16* r1BF   = (__hip_bfloat16*)carve(MTOK * 128 * 2);
  __hip_bfloat16* prBF   = (__hip_bfloat16*)carve(MTOK * 32 * 2);
  float*          simF   = (float*)carve((long long)BB * TT * TT * 4);
  __hip_bfloat16* xbBF   = (__hip_bfloat16*)carve(MTOK * DD * 2);
  __hip_bfloat16* embT   = (__hip_bfloat16*)carve((long long)VV * DD * 2);
  __hip_bfloat16* wqT    = (__hip_bfloat16*)carve((long long)DD * DD * 2);
  __hip_bfloat16* wkT    = (__hip_bfloat16*)carve((long long)DD * DD * 2);
  __hip_bfloat16* wvT    = (__hip_bfloat16*)carve((long long)DD * DD * 2);
  __hip_bfloat16* woT    = (__hip_bfloat16*)carve((long long)DD * DD * 2);
  __hip_bfloat16* rw1T   = (__hip_bfloat16*)carve((long long)DD * 128 * 2);
  __hip_bfloat16* mw1T   = (__hip_bfloat16*)carve((long long)DD * DFF * 2);
  __hip_bfloat16* mw2T   = (__hip_bfloat16*)carve((long long)DFF * DD * 2);

  auto gemm = [&](const __hip_bfloat16* A, const __hip_bfloat16* Wt,
                  const float* bias, const float* res, float* oF,
                  __hip_bfloat16* oB, int M, int N, int K, int epi, int batch,
                  long long sA, long long sW, long long sC) {
    dim3 g(N / 128, M / 128, batch);
    gemm_bf16_wmma<<<g, dim3(256), 0, stream>>>(A, Wt, bias, res, oF, oB,
                                                M, N, K, epi, sA, sW, sC);
  };
  auto tcvt = [&](const float* in, __hip_bfloat16* o, int R, int C) {
    tcvt_k<<<dim3(C / 32, R / 32), dim3(32, 8), 0, stream>>>(in, o, R, C);
  };

  // embedding + lm_head weight conversion (token_emb [V,D] is already [N,K])
  {
    const long long n = MTOK * DD;
    embed_k<<<(int)((n + 255) / 256), 256, 0, stream>>>(tokens, token_emb, pos_emb, xF);
    const long long ne = (long long)VV * DD;
    cvt_bf16_k<<<(int)((ne + 255) / 256), 256, 0, stream>>>(token_emb, embT, ne);
  }

  for (int l = 0; l < LL; ++l) {
    // per-layer weight transpose+convert into reused buffers
    tcvt(Wq + (long long)l * DD * DD, wqT, DD, DD);
    tcvt(Wk + (long long)l * DD * DD, wkT, DD, DD);
    tcvt(Wv + (long long)l * DD * DD, wvT, DD, DD);
    tcvt(Wo + (long long)l * DD * DD, woT, DD, DD);
    tcvt(rw1 + (long long)l * DD * 128, rw1T, DD, 128);
    tcvt(mw1 + (long long)l * DD * DFF, mw1T, DD, DFF);
    tcvt(mw2 + (long long)l * DFF * DD, mw2T, DFF, DD);

    // h = rms(x, n1w)
    rmsnorm_bf16<<<(int)MTOK, 256, 0, stream>>>(xF, n1w + (long long)l * DD, hBF);

    // q, kf, v projections
    gemm(hBF, wqT, nullptr, nullptr, nullptr, qBF, (int)MTOK, DD, DD, 0, 1, 0, 0, 0);
    gemm(hBF, wkT, nullptr, nullptr, nullptr, kBF, (int)MTOK, DD, DD, 0, 1, 0, 0, 0);
    gemm(hBF, wvT, nullptr, nullptr, nullptr, vBF, (int)MTOK, DD, DD, 0, 1, 0, 0, 0);

    // routing: r1 = tanh(kf @ rw1 + rb1); probs = softmax(r1 @ rw2 + rb2)
    gemm(kBF, rw1T, rb1 + (long long)l * 128, nullptr, nullptr, r1BF,
         (int)MTOK, 128, DD, 2, 1, 0, 0, 0);
    routing_k<<<(int)(MTOK / 16), 256, 0, stream>>>(
        r1BF, rw2 + (long long)l * 128 * NBK, rb2 + (long long)l * NBK, prBF);

    // sim = probs @ probs^T  (batched over B, K padded 16->32)
    gemm(prBF, prBF, nullptr, nullptr, simF, nullptr, TT, TT, 32, 0, BB,
         (long long)TT * 32, (long long)TT * 32, (long long)TT * TT);

    // V^T per head for PV wmma
    {
      const long long n = MTOK * DD;
      vtrans_k<<<(int)((n + 255) / 256), 256, 0, stream>>>(vBF, vTbuf);
    }

    // attention
    attn_wmma<<<dim3(TT / 64, BB * HH), dim3(128), 0, stream>>>(
        qBF, kBF, vTbuf, simF, oBF);

    // x = x + o @ Wo
    gemm(oBF, woT, nullptr, xF, xF, nullptr, (int)MTOK, DD, DD, 0, 1, 0, 0, 0);

    // MLP
    rmsnorm_bf16<<<(int)MTOK, 256, 0, stream>>>(xF, n2w + (long long)l * DD, hBF);
    gemm(hBF, mw1T, mb1 + (long long)l * DFF, nullptr, nullptr, midBF,
         (int)MTOK, DFF, DD, 1, 1, 0, 0, 0);
    gemm(midBF, mw2T, mb2 + (long long)l * DD, xF, xF, nullptr,
         (int)MTOK, DD, DFF, 0, 1, 0, 0, 0);
  }

  // logits = x @ token_emb^T
  {
    const long long n = MTOK * DD;
    cvt_bf16_k<<<(int)((n + 255) / 256), 256, 0, stream>>>(xF, xbBF, n);
    gemm(xbBF, embT, nullptr, nullptr, out, nullptr, (int)MTOK, VV, DD, 0, 1, 0, 0, 0);
  }
}
